// SEEDTransformer_24464133718851
// MI455X (gfx1250) — matrix operand order repaired and checked by reading, not compiled, not run on hardware
//
#include <hip/hip_runtime.h>
#include <math.h>

// ---------------- problem constants ----------------
#define BB    2
#define LL    35344
#define DD    256
#define NHH   8
#define HDD   32
#define PPK   25
#define FGG   7069
#define QNN   1414
#define DFFF  1024
#define WII   188

// ---------------- WMMA types / helpers ----------------
typedef __attribute__((ext_vector_type(16))) __bf16 v16bf;
typedef __attribute__((ext_vector_type(8)))  __bf16 v8bf;
typedef __attribute__((ext_vector_type(8)))  float  v8f;

__device__ __forceinline__ unsigned short f2bfu(float f) {
  unsigned u = __builtin_bit_cast(unsigned, f);
  u += 0x7FFFu + ((u >> 16) & 1u);          // round-to-nearest-even
  return (unsigned short)(u >> 16);
}
__device__ __forceinline__ __bf16 bf_from_u16(unsigned short h) {
  return __builtin_bit_cast(__bf16, h);
}
__device__ __forceinline__ __bf16 f2bf(float f) { return bf_from_u16(f2bfu(f)); }

__device__ __forceinline__ v8f wmma_bf16(v16bf a, v16bf b, v8f c) {
  // D = A(16x32 bf16) x B(32x16 bf16) + C(16x16 f32)
  return __builtin_amdgcn_wmma_f32_16x16x32_bf16(false, a, false, b, (short)0, c,
                                                 false, false);
}
__device__ __forceinline__ v16bf cat8(v8bf lo, v8bf hi) {
  return __builtin_shufflevector(lo, hi, 0, 1, 2, 3, 4, 5, 6, 7,
                                 8, 9, 10, 11, 12, 13, 14, 15);
}

// ---------------- f32 -> bf16 conversion pass ----------------
__global__ void cvt_f32_bf16_kernel(const float* __restrict__ a, __bf16* __restrict__ o, size_t n) {
  size_t i = (size_t)blockIdx.x * 256 + threadIdx.x;
  if (i < n) o[i] = f2bf(a[i]);
}

// ---------------- top-k by exact rank ----------------
__global__ void topk_rank_kernel(const float* __restrict__ score, int* __restrict__ idx) {
  int b = blockIdx.y;
  int i = blockIdx.x * 256 + threadIdx.x;
  const float* s = score + (size_t)b * LL;
  float si = (i < LL) ? s[i] : -3e38f;
  __shared__ float tile[256];
  int rank = 0;
  for (int j0 = 0; j0 < LL; j0 += 256) {
    int jt = j0 + threadIdx.x;
    tile[threadIdx.x] = (jt < LL) ? s[jt] : -3e38f;
    __syncthreads();
    int lim = LL - j0; if (lim > 256) lim = 256;
    for (int jj = 0; jj < lim; ++jj) {
      float sj = tile[jj];
      int j = j0 + jj;
      rank += ((sj > si) || (sj == si && j < i)) ? 1 : 0;
    }
    __syncthreads();
  }
  if (i < LL && rank < FGG) idx[(size_t)b * FGG + rank] = i;
}

// ---------------- generic row gather: dst[b,r,:] = src[b, idx[b,r], :] ----------------
__global__ void gather_rows_kernel(const float* __restrict__ srcM, const int* __restrict__ idx,
                                   float* __restrict__ dst, int cols) {
  size_t i = (size_t)blockIdx.x * 256 + threadIdx.x;
  size_t total = (size_t)BB * FGG * cols;
  if (i >= total) return;
  int c = (int)(i % cols);
  int r = (int)(i / cols);
  int b = r / FGG;
  int row = idx[r];
  if (row < 0) row = 0; if (row >= LL) row = LL - 1;
  dst[i] = srcM[((size_t)b * LL + row) * cols + c];
}

// ---- build q = sel_src[:QN]+sel_pos[:QN] (bf16), qsrc (bf16 + f32 for residual) ----
__global__ void build_q_kernel(const float* __restrict__ sel_src, const float* __restrict__ sel_pos,
                               __bf16* __restrict__ qbuf_bf, __bf16* __restrict__ qsrc_bf,
                               float* __restrict__ qsrc_f) {
  size_t i = (size_t)blockIdx.x * 256 + threadIdx.x;
  size_t total = (size_t)BB * QNN * DD;
  if (i >= total) return;
  int c = (int)(i % DD);
  int r = (int)(i / DD);
  int b = r / QNN;
  int f = r % QNN;
  size_t srow = ((size_t)b * FGG + f) * DD + c;
  float s = sel_src[srow];
  qsrc_f[i] = s;
  qsrc_bf[i] = f2bf(s);
  qbuf_bf[i] = f2bf(s + sel_pos[srow]);
}

// ---------------- scatter LN'd query rows back into sel_src[:QN] ----------------
__global__ void scatter_qn_kernel(const float* __restrict__ lnq, float* __restrict__ sel_src) {
  size_t i = (size_t)blockIdx.x * 256 + threadIdx.x;
  size_t total = (size_t)BB * QNN * DD;
  if (i >= total) return;
  int c = (int)(i % DD);
  int r = (int)(i / DD);
  int b = r / QNN;
  int f = r % QNN;
  sel_src[((size_t)b * FGG + f) * DD + c] = lnq[i];
}

// ---------------- elementwise helpers ----------------
__global__ void add_bf_kernel(const float* __restrict__ a, const float* __restrict__ b,
                              __bf16* __restrict__ c, size_t n) {
  size_t i = (size_t)blockIdx.x * 256 + threadIdx.x;
  if (i < n) c[i] = f2bf(a[i] + b[i]);
}
__global__ void copy_kernel(const float* __restrict__ a, float* __restrict__ b, size_t n) {
  size_t i = (size_t)blockIdx.x * 256 + threadIdx.x;
  if (i < n) b[i] = a[i];
}

// ---- bf16 WMMA GEMM: C(MxN) = A(MxK)bf16 @ W(NxK)bf16^T + bias (+resid)(+relu) ----
// Optional f32 and/or bf16 outputs. block (32,4): each wave does a 16x64 tile.
__global__ void gemm_bf16_wmma_kernel(const __bf16* __restrict__ A, const __bf16* __restrict__ W,
                                      const float* __restrict__ bias, const float* __restrict__ resid,
                                      float* __restrict__ Cf, __bf16* __restrict__ Cb,
                                      int M, int N, int K, int relu) {
  const int lane = threadIdx.x;
  const int wave = threadIdx.y;
  const int m0 = (blockIdx.x * 4 + wave) * 16;
  if (m0 >= M) return;
  const int n0 = blockIdx.y * 64;
  const int half = lane >> 4;
  const int ml = lane & 15;
  int mrow = m0 + ml; if (mrow >= M) mrow = M - 1;

  const __bf16* ap = A + (size_t)mrow * K + half * 16;
  int nidx[4];
#pragma unroll
  for (int t = 0; t < 4; ++t) {
    int n = n0 + t * 16 + ml;
    nidx[t] = (n < N) ? n : (N - 1);
  }
  const __bf16* wp0 = W + (size_t)nidx[0] * K + half * 16;
  const __bf16* wp1 = W + (size_t)nidx[1] * K + half * 16;
  const __bf16* wp2 = W + (size_t)nidx[2] * K + half * 16;
  const __bf16* wp3 = W + (size_t)nidx[3] * K + half * 16;

  v8f acc[4] = {};
  for (int k0 = 0; k0 < K; k0 += 32) {
    v16bf a = cat8(*(const v8bf*)(ap + k0), *(const v8bf*)(ap + k0 + 8));
    __builtin_prefetch((const void*)(ap + k0 + 32), 0, 1);   // global_prefetch_b8
    acc[0] = wmma_bf16(a, *(const v16bf*)(wp0 + k0), acc[0]);
    acc[1] = wmma_bf16(a, *(const v16bf*)(wp1 + k0), acc[1]);
    acc[2] = wmma_bf16(a, *(const v16bf*)(wp2 + k0), acc[2]);
    acc[3] = wmma_bf16(a, *(const v16bf*)(wp3 + k0), acc[3]);
  }
#pragma unroll
  for (int t = 0; t < 4; ++t) {
    int n = n0 + t * 16 + ml;
    if (n >= N) continue;
    float bb = bias ? bias[n] : 0.0f;
#pragma unroll
    for (int v = 0; v < 8; ++v) {
      int m = m0 + v + half * 8;
      if (m >= M) continue;
      float val = acc[t][v] + bb;
      if (resid) val += resid[(size_t)m * N + n];
      if (relu) val = fmaxf(val, 0.0f);
      if (Cf) Cf[(size_t)m * N + n] = val;
      if (Cb) Cb[(size_t)m * N + n] = f2bf(val);
    }
  }
}

// ---------------- flash attention (per head, HD=32), WMMA for QK^T and PV -------------
// qp/kp/vp: bf16 (B*QN, 256); out: bf16. grid: (ceil(numqt/4), NH, B); block (32,4).
__global__ void attn_kernel(const __bf16* __restrict__ qp, const __bf16* __restrict__ kp,
                            const __bf16* __restrict__ vp, __bf16* __restrict__ outp) {
  __shared__ unsigned short Pld[4][16 * 32];
  const int lane = threadIdx.x, wv = threadIdx.y;
  const int numqt = (QNN + 15) / 16;
  const int qt = blockIdx.x * 4 + wv;
  const int h = blockIdx.y, b = blockIdx.z;
  const bool active = (qt < numqt);
  const int half = lane >> 4, nl = lane & 15;
  const int m0 = qt * 16;
  const size_t base = (size_t)b * QNN * DD + (size_t)h * HDD;
  const float scl = 0.17677669529663687f; // 1/sqrt(32)

  v16bf aQ;
  if (active) {
    int q = m0 + nl; if (q >= QNN) q = QNN - 1;
    const __bf16* qrow = qp + base + (size_t)q * DD + half * 16;
    aQ = cat8(*(const v8bf*)qrow, *(const v8bf*)(qrow + 8));
  }
  float rmax[8], rsum[8];
#pragma unroll
  for (int v = 0; v < 8; ++v) { rmax[v] = -3e38f; rsum[v] = 0.0f; }
  v8f O0 = {}, O1 = {};

  const int nchunks = (QNN + 31) / 32;
  for (int ck = 0; ck < nchunks; ++ck) {
    const int kc = ck * 32;
    if (active) {
      int key0 = kc + nl, key1 = kc + 16 + nl;
      int k0c = key0 < QNN ? key0 : QNN - 1;
      int k1c = key1 < QNN ? key1 : QNN - 1;
      v16bf b0 = *(const v16bf*)(kp + base + (size_t)k0c * DD + half * 16);
      v16bf b1 = *(const v16bf*)(kp + base + (size_t)k1c * DD + half * 16);
      v8f S0 = {}, S1 = {};
      S0 = wmma_bf16(aQ, b0, S0);
      S1 = wmma_bf16(aQ, b1, S1);
      const bool ok0 = key0 < QNN, ok1 = key1 < QNN;
#pragma unroll
      for (int v = 0; v < 8; ++v) {
        S0[v] = ok0 ? S0[v] * scl : -3e38f;
        S1[v] = ok1 ? S1[v] * scl : -3e38f;
      }
#pragma unroll
      for (int v = 0; v < 8; ++v) {
        float cm = fmaxf(S0[v], S1[v]);
        cm = fmaxf(cm, __shfl_xor(cm, 8));
        cm = fmaxf(cm, __shfl_xor(cm, 4));
        cm = fmaxf(cm, __shfl_xor(cm, 2));
        cm = fmaxf(cm, __shfl_xor(cm, 1));
        float nm = fmaxf(rmax[v], cm);
        float corr = expf(rmax[v] - nm);
        rmax[v] = nm;
        float e0 = expf(S0[v] - nm);
        float e1 = expf(S1[v] - nm);
        float cs = e0 + e1;
        cs += __shfl_xor(cs, 8);
        cs += __shfl_xor(cs, 4);
        cs += __shfl_xor(cs, 2);
        cs += __shfl_xor(cs, 1);
        rsum[v] = rsum[v] * corr + cs;
        O0[v] *= corr;
        O1[v] *= corr;
        int mrow = v + half * 8;
        Pld[wv][mrow * 32 + nl] = f2bfu(e0);
        Pld[wv][mrow * 32 + 16 + nl] = f2bfu(e1);
      }
    }
    __syncthreads();
    if (active) {
      v16bf aP, bV0, bV1;
#pragma unroll
      for (int c = 0; c < 8; ++c) aP[c] = bf_from_u16(Pld[wv][nl * 32 + half * 8 + c]);
#pragma unroll
      for (int c = 0; c < 8; ++c) aP[8 + c] = bf_from_u16(Pld[wv][nl * 32 + 16 + half * 8 + c]);
#pragma unroll
      for (int e = 0; e < 16; ++e) {
        int key = kc + (e < 8 ? half * 8 + e : 16 + half * 8 + (e - 8));
        if (key >= QNN) key = QNN - 1;
        const __bf16* vr = vp + base + (size_t)key * DD;
        bV0[e] = vr[nl];
        bV1[e] = vr[16 + nl];
      }
      O0 = wmma_bf16(aP, bV0, O0);
      O1 = wmma_bf16(aP, bV1, O1);
    }
    __syncthreads();
  }
  if (active) {
#pragma unroll
    for (int v = 0; v < 8; ++v) {
      int m = m0 + v + half * 8;
      if (m >= QNN) continue;
      float inv = 1.0f / rsum[v];
      outp[base + (size_t)m * DD + nl] = f2bf(O0[v] * inv);
      outp[base + (size_t)m * DD + 16 + nl] = f2bf(O1[v] * inv);
    }
  }
}

// ------- LayerNorm over D=256, optional residual input + optional bf16 mirror out -------
// block (32,8): one wave per row
__global__ void ln_kernel(const float* __restrict__ x, const float* __restrict__ res,
                          const float* __restrict__ g, const float* __restrict__ beta,
                          float* __restrict__ outp, __bf16* __restrict__ outb, int rows) {
  int row = blockIdx.x * 8 + threadIdx.y;
  if (row >= rows) return;
  int lane = threadIdx.x;
  const float* xr = x + (size_t)row * DD;
  const float* rr = res ? res + (size_t)row * DD : nullptr;
  float vals[8];
  float s = 0.0f;
#pragma unroll
  for (int c = 0; c < 8; ++c) {
    int j = lane * 8 + c;
    float v = xr[j] + (rr ? rr[j] : 0.0f);
    vals[c] = v;
    s += v;
  }
  for (int m = 16; m >= 1; m >>= 1) s += __shfl_xor(s, m);
  float mean = s * (1.0f / DD);
  float sq = 0.0f;
#pragma unroll
  for (int c = 0; c < 8; ++c) { float d = vals[c] - mean; sq += d * d; }
  for (int m = 16; m >= 1; m >>= 1) sq += __shfl_xor(sq, m);
  float rstd = rsqrtf(sq * (1.0f / DD) + 1e-5f);
#pragma unroll
  for (int c = 0; c < 8; ++c) {
    int j = lane * 8 + c;
    float y = (vals[c] - mean) * rstd * g[j] + beta[j];
    outp[(size_t)row * DD + j] = y;
    if (outb) outb[(size_t)row * DD + j] = f2bf(y);
  }
}

// ---------------- grid points + aw softmax (in-place over sop / awp) -------------------
__global__ void grid_aw_kernel(const float* __restrict__ selref, const float* __restrict__ boxp,
                               float* __restrict__ sop, float* __restrict__ awp) {
  int t = blockIdx.x * 256 + threadIdx.x;
  int total = BB * FGG * NHH;
  if (t >= total) return;
  int h = t % NHH;
  int q = t / NHH;
  const float* rw = selref + (size_t)q * 7;
  const float* ob = boxp + (size_t)q * (NHH * 4) + h * 4;
  float cx = rw[0] + ob[0] * 0.125f * rw[3];
  float cy = rw[1] + ob[1] * 0.125f * rw[4];
  float w  = fmaxf(rw[3] + ob[2] * 0.125f * rw[3], 0.0f);
  float hh = fmaxf(rw[4] + ob[3] * 0.125f * rw[4], 0.0f);
  float ca = cosf(rw[6]);
  float sa = sinf(rw[6]);
  float* sp = sop + (size_t)q * (NHH * PPK * 2) + (size_t)h * (PPK * 2);
  for (int p = 0; p < PPK; ++p) {
    float kx = (float)((p % 5) - 2) * 0.2f;
    float ky = (float)((p / 5) - 2) * 0.2f;
    float fx = kx * w, fy = ky * hh;
    float rx = ca * fx - sa * fy;
    float ry = sa * fx + ca * fy;
    sp[p * 2 + 0] = cx + rx + sp[p * 2 + 0] * (1.0f / 188.0f);
    sp[p * 2 + 1] = cy + ry + sp[p * 2 + 1] * (1.0f / 188.0f);
  }
  float* apw = awp + (size_t)q * (NHH * PPK) + (size_t)h * PPK;
  float mx = -3e38f;
  for (int p = 0; p < PPK; ++p) mx = fmaxf(mx, apw[p]);
  float e[PPK];
  float ssum = 0.0f;
  for (int p = 0; p < PPK; ++p) { e[p] = expf(apw[p] - mx); ssum += e[p]; }
  float inv = 1.0f / ssum;
  for (int p = 0; p < PPK; ++p) apw[p] = e[p] * inv;
}

// ---------------- bilinear sample + attention-weighted sum (bf16 out) -------------------
// grid: B*FG blocks; block (32,8): lane=d, y=h
__global__ void sample_kernel(const float* __restrict__ value, const float* __restrict__ gridp,
                              const float* __restrict__ aw, __bf16* __restrict__ outp) {
  int q = blockIdx.x;            // b*FG + f
  int d = threadIdx.x;
  int h = threadIdx.y;
  int b = q / FGG;
  const float* gp = gridp + ((size_t)q * NHH + h) * (PPK * 2);
  const float* ap = aw + ((size_t)q * NHH + h) * PPK;
  const float* vb = value + (size_t)b * LL * DD + (size_t)h * HDD + d;
  float acc = 0.0f;
  for (int p = 0; p < PPK; ++p) {
    float px = gp[p * 2 + 0] * (float)WII - 0.5f;
    float py = gp[p * 2 + 1] * (float)WII - 0.5f;
    float x0f = floorf(px), y0f = floorf(py);
    float lx = px - x0f, ly = py - y0f;
    int x0 = (int)x0f, y0 = (int)y0f;
    float wp = ap[p];
    float w00 = wp * (1.0f - lx) * (1.0f - ly);
    float w10 = wp * lx * (1.0f - ly);
    float w01 = wp * (1.0f - lx) * ly;
    float w11 = wp * lx * ly;
    int x1 = x0 + 1, y1 = y0 + 1;
    if (x0 >= 0 && x0 < WII && y0 >= 0 && y0 < WII) acc += w00 * vb[(size_t)(y0 * WII + x0) * DD];
    if (x1 >= 0 && x1 < WII && y0 >= 0 && y0 < WII) acc += w10 * vb[(size_t)(y0 * WII + x1) * DD];
    if (x0 >= 0 && x0 < WII && y1 >= 0 && y1 < WII) acc += w01 * vb[(size_t)(y1 * WII + x0) * DD];
    if (x1 >= 0 && x1 < WII && y1 >= 0 && y1 < WII) acc += w11 * vb[(size_t)(y1 * WII + x1) * DD];
  }
  outp[(size_t)q * DD + (size_t)h * HDD + d] = f2bf(acc);
}

// ---------------- final scatter: out[b, idx[b,r], :] = sel_src[b, r, :] -----------------
__global__ void scatter_out_kernel(const float* __restrict__ sel, const int* __restrict__ idx,
                                   float* __restrict__ outp) {
  size_t i = (size_t)blockIdx.x * 256 + threadIdx.x;
  size_t total = (size_t)BB * FGG * DD;
  if (i >= total) return;
  int c = (int)(i % DD);
  int r = (int)(i / DD);
  int b = r / FGG;
  int row = idx[r];
  if (row < 0) row = 0; if (row >= LL) row = LL - 1;
  outp[((size_t)b * LL + row) * DD + c] = sel[i];
}

// ======================= host side =======================
static void launch_gemm(const __bf16* A, const __bf16* W, const float* bias, const float* resid,
                        float* Cf, __bf16* Cb, int M, int N, int K, int relu, hipStream_t stream) {
  dim3 g((unsigned)((M + 63) / 64), (unsigned)((N + 63) / 64));
  dim3 blk(32, 4);
  gemm_bf16_wmma_kernel<<<g, blk, 0, stream>>>(A, W, bias, resid, Cf, Cb, M, N, K, relu);
}
static void launch_cvt(const float* a, __bf16* o, size_t n, hipStream_t stream) {
  cvt_f32_bf16_kernel<<<(unsigned)((n + 255) / 256), 256, 0, stream>>>(a, o, n);
}

extern "C" void kernel_launch(void* const* d_in, const int* in_sizes, int n_in,
                              void* d_out, int out_size, void* d_ws, size_t ws_size,
                              hipStream_t stream) {
  (void)in_sizes; (void)n_in; (void)out_size; (void)ws_size;
  const float* src       = (const float*)d_in[0];
  const float* pos       = (const float*)d_in[1];
  const float* refw      = (const float*)d_in[2];
  const float* score     = (const float*)d_in[3];
  const float* mha_in_w  = (const float*)d_in[4];
  const float* mha_in_b  = (const float*)d_in[5];
  const float* mha_out_w = (const float*)d_in[6];
  const float* mha_out_b = (const float*)d_in[7];
  const float* value_w   = (const float*)d_in[8];
  const float* value_b   = (const float*)d_in[9];
  const float* ca_out_w  = (const float*)d_in[10];
  const float* ca_out_b  = (const float*)d_in[11];
  const float* box_w     = (const float*)d_in[12];
  const float* box_b     = (const float*)d_in[13];
  const float* aw_w      = (const float*)d_in[14];
  const float* aw_b      = (const float*)d_in[15];
  const float* so_w      = (const float*)d_in[16];
  const float* so_b      = (const float*)d_in[17];
  const float* lin1_w    = (const float*)d_in[18];
  const float* lin1_b    = (const float*)d_in[19];
  const float* lin2_w    = (const float*)d_in[20];
  const float* lin2_b    = (const float*)d_in[21];
  const float* qn_g      = (const float*)d_in[22];
  const float* qn_bv     = (const float*)d_in[23];
  const float* n1_g      = (const float*)d_in[24];
  const float* n1_b      = (const float*)d_in[25];
  const float* n2_g      = (const float*)d_in[26];
  const float* n2_b      = (const float*)d_in[27];
  float* outp = (float*)d_out;

  // ---- workspace carve ----
  char* wsb = (char*)d_ws;
  size_t off = 0;
  auto allocB = [&](size_t bytes) -> void* {
    void* p = (void*)(wsb + off);
    off += ((bytes + 255) / 256) * 256;
    return p;
  };
  auto allocF = [&](size_t n) -> float* { return (float*)allocB(n * 4); };
  auto allocH = [&](size_t n) -> __bf16* { return (__bf16*)allocB(n * 2); };

  int*    idx      = (int*)allocB((size_t)BB * FGG * 4);
  float*  sel_src  = allocF((size_t)BB * FGG * DD);
  float*  sel_pos  = allocF((size_t)BB * FGG * DD);
  float*  sel_ref  = allocF((size_t)BB * FGG * 7);
  __bf16* qbuf_bf  = allocH((size_t)BB * QNN * DD);
  __bf16* qsrc_bf  = allocH((size_t)BB * QNN * DD);
  float*  qsrc_f   = allocF((size_t)BB * QNN * DD);
  __bf16* qp_bf    = allocH((size_t)BB * QNN * DD);
  __bf16* kp_bf    = allocH((size_t)BB * QNN * DD);
  __bf16* vp_bf    = allocH((size_t)BB * QNN * DD);
  __bf16* attn_bf  = allocH((size_t)BB * QNN * DD);
  float*  mha      = allocF((size_t)BB * QNN * DD);
  float*  lnq      = allocF((size_t)BB * QNN * DD);
  __bf16* src_bf   = allocH((size_t)BB * LL * DD);
  float*  value    = allocF((size_t)BB * LL * DD);   // later aliased by ffn1_bf
  __bf16* query2_bf= allocH((size_t)BB * FGG * DD);
  float*  boxp     = allocF((size_t)BB * FGG * NHH * 4);
  float*  sop      = allocF((size_t)BB * FGG * NHH * PPK * 2); // becomes grid pts in-place
  float*  awp      = allocF((size_t)BB * FGG * NHH * PPK);     // softmaxed in-place
  __bf16* sampo_bf = allocH((size_t)BB * FGG * DD);
  float*  cab      = allocF((size_t)BB * FGG * DD);
  __bf16* sel_bf   = allocH((size_t)BB * FGG * DD);
  float*  ffn2     = allocF((size_t)BB * FGG * DD);
  __bf16* ffn1_bf  = (__bf16*)value;                 // reuse (value consumed by sampling)

  // bf16 weight mirrors
  __bf16* w_mha_in = allocH((size_t)3 * DD * DD);
  __bf16* w_mha_out= allocH((size_t)DD * DD);
  __bf16* w_value  = allocH((size_t)DD * DD);
  __bf16* w_ca     = allocH((size_t)DD * DD);
  __bf16* w_box    = allocH((size_t)NHH * 4 * DD);
  __bf16* w_so     = allocH((size_t)NHH * PPK * 2 * DD);
  __bf16* w_aw     = allocH((size_t)NHH * PPK * DD);
  __bf16* w_lin1   = allocH((size_t)DFFF * DD);
  __bf16* w_lin2   = allocH((size_t)DD * DFFF);

  const int MQ = BB * QNN;   // 2828
  const int MF = BB * FGG;   // 14138
  const int MV = BB * LL;    // 70688

  // 0. one-time f32 -> bf16 conversions (weights + src)
  launch_cvt(mha_in_w,  w_mha_in,  (size_t)3 * DD * DD, stream);
  launch_cvt(mha_out_w, w_mha_out, (size_t)DD * DD, stream);
  launch_cvt(value_w,   w_value,   (size_t)DD * DD, stream);
  launch_cvt(ca_out_w,  w_ca,      (size_t)DD * DD, stream);
  launch_cvt(box_w,     w_box,     (size_t)NHH * 4 * DD, stream);
  launch_cvt(so_w,      w_so,      (size_t)NHH * PPK * 2 * DD, stream);
  launch_cvt(aw_w,      w_aw,      (size_t)NHH * PPK * DD, stream);
  launch_cvt(lin1_w,    w_lin1,    (size_t)DFFF * DD, stream);
  launch_cvt(lin2_w,    w_lin2,    (size_t)DD * DFFF, stream);
  launch_cvt(src,       src_bf,    (size_t)MV * DD, stream);

  // 1. top-k selection (exact rank)
  topk_rank_kernel<<<dim3((LL + 255) / 256, BB), 256, 0, stream>>>(score, idx);

  // 2. gathers
  {
    size_t n = (size_t)MF * DD;
    unsigned g = (unsigned)((n + 255) / 256);
    gather_rows_kernel<<<g, 256, 0, stream>>>(src, idx, sel_src, DD);
    gather_rows_kernel<<<g, 256, 0, stream>>>(pos, idx, sel_pos, DD);
    size_t n7 = (size_t)MF * 7;
    gather_rows_kernel<<<(unsigned)((n7 + 255) / 256), 256, 0, stream>>>(refw, idx, sel_ref, 7);
  }

  // 3. q / qsrc (q_idx == arange(QN): sel_score already sorted descending)
  {
    size_t n = (size_t)MQ * DD;
    build_q_kernel<<<(unsigned)((n + 255) / 256), 256, 0, stream>>>(sel_src, sel_pos,
                                                                    qbuf_bf, qsrc_bf, qsrc_f);
  }

  // 4. QKV projections -> bf16
  launch_gemm(qbuf_bf, w_mha_in + 0 * DD * DD, mha_in_b + 0 * DD, nullptr, nullptr, qp_bf, MQ, DD, DD, 0, stream);
  launch_gemm(qbuf_bf, w_mha_in + 1 * DD * DD, mha_in_b + 1 * DD, nullptr, nullptr, kp_bf, MQ, DD, DD, 0, stream);
  launch_gemm(qsrc_bf, w_mha_in + 2 * DD * DD, mha_in_b + 2 * DD, nullptr, nullptr, vp_bf, MQ, DD, DD, 0, stream);

  // 5. flash attention per head (bf16 in/out)
  {
    int numqt = (QNN + 15) / 16;
    dim3 g((unsigned)((numqt + 3) / 4), NHH, BB);
    attn_kernel<<<g, dim3(32, 4), 0, stream>>>(qp_bf, kp_bf, vp_bf, attn_bf);
  }

  // 6. out projection + residual -> f32; 7. qn LayerNorm; 8. scatter into sel_src[:QN]
  launch_gemm(attn_bf, w_mha_out, mha_out_b, qsrc_f, mha, nullptr, MQ, DD, DD, 0, stream);
  ln_kernel<<<(unsigned)((MQ + 7) / 8), dim3(32, 8), 0, stream>>>(mha, nullptr, qn_g, qn_bv, lnq, nullptr, MQ);
  {
    size_t n = (size_t)MQ * DD;
    scatter_qn_kernel<<<(unsigned)((n + 255) / 256), 256, 0, stream>>>(lnq, sel_src);
  }

  // 9. value projection over all L (biggest GEMM, bf16 in -> f32 out)
  launch_gemm(src_bf, w_value, value_b, nullptr, value, nullptr, MV, DD, DD, 0, stream);

  // 10. query2 = sel_src + sel_pos -> bf16
  {
    size_t n = (size_t)MF * DD;
    add_bf_kernel<<<(unsigned)((n + 255) / 256), 256, 0, stream>>>(sel_src, sel_pos, query2_bf, n);
  }

  // 11-13. box / sampling-offset / attention-weight projections
  launch_gemm(query2_bf, w_box, box_b, nullptr, boxp, nullptr, MF, NHH * 4, DD, 0, stream);
  launch_gemm(query2_bf, w_so,  so_b,  nullptr, sop,  nullptr, MF, NHH * PPK * 2, DD, 0, stream);
  launch_gemm(query2_bf, w_aw,  aw_b,  nullptr, awp,  nullptr, MF, NHH * PPK, DD, 0, stream);

  // 14. grid points + aw softmax (in-place)
  {
    int total = MF * NHH;
    grid_aw_kernel<<<(unsigned)((total + 255) / 256), 256, 0, stream>>>(sel_ref, boxp, sop, awp);
  }

  // 15. bilinear sampling + weighted sum -> bf16
  sample_kernel<<<(unsigned)MF, dim3(32, 8), 0, stream>>>(value, sop, awp, sampo_bf);

  // 16. ca_out projection; n1 LayerNorm (bf16 mirror feeds FFN)
  launch_gemm(sampo_bf, w_ca, ca_out_b, nullptr, cab, nullptr, MF, DD, DD, 0, stream);
  ln_kernel<<<(unsigned)((MF + 7) / 8), dim3(32, 8), 0, stream>>>(sel_src, cab, n1_g, n1_b, sel_src, sel_bf, MF);

  // 17-18. FFN (hidden kept in bf16, reuses value buffer); 19. n2 LayerNorm
  launch_gemm(sel_bf, w_lin1, lin1_b, nullptr, nullptr, ffn1_bf, MF, DFFF, DD, 1, stream);
  launch_gemm(ffn1_bf, w_lin2, lin2_b, nullptr, ffn2, nullptr, MF, DD, DFFF, 0, stream);
  ln_kernel<<<(unsigned)((MF + 7) / 8), dim3(32, 8), 0, stream>>>(sel_src, ffn2, n2_g, n2_b, sel_src, nullptr, MF);

  // 20. copy src -> out, 21. scatter updated rows
  {
    size_t n = (size_t)MV * DD;
    copy_kernel<<<(unsigned)((n + 255) / 256), 256, 0, stream>>>(src, outp, n);
    size_t ns = (size_t)MF * DD;
    scatter_out_kernel<<<(unsigned)((ns + 255) / 256), 256, 0, stream>>>(sel_src, idx, outp);
  }
}
